// EmotionCap_41068477285252
// MI455X (gfx1250) — matrix operand order repaired and checked by reading, not compiled, not run on hardware
//
#include <hip/hip_runtime.h>
#include <hip/hip_bf16.h>

// Capsule routing on MI455X (gfx1250, wave32).
// B=64 batch, N=16 capsules, I=2048 input capsules, D=K=32, 3 routing rounds.
//
// Stage 1: u_hat[b,i,n,d] = sum_k W[n,i,d,k] * x[b,i,k]   (bf16 WMMA, f32 acc)
//          W slice per i staged into LDS by the Tensor Data Mover (TDM) with
//          bank-conflict-avoiding padding; u_hat stored bf16, layout [B,I,N,D].
// Stage 2: per round r: ONE fused pass over u_hat computes (for r>0) the
//          agreement a, logit update, softmax over n, and accumulates
//          s[b,n,d] = sum_i c*u via register accum -> LDS atomics -> global atomics.
// Stage 3: tiny squash kernel produces v (final round -> d_out).

#define BATCH 64
#define NCAPS 16
#define ICAPS 2048
#define DDIM  32
#define KDIM  32
#define EPSF  1e-7f

typedef __attribute__((ext_vector_type(16))) __bf16 v16bf;
typedef __attribute__((ext_vector_type(8)))  float  v8f;
typedef __attribute__((ext_vector_type(4)))  unsigned int u32x4;
typedef __attribute__((ext_vector_type(8)))  int i32x8;
typedef __attribute__((ext_vector_type(4)))  int i32x4;

// LDS W slice: 16 n-rows x 32 d-rows, each d-row 32 floats + 4 floats TDM pad
#define WROW 36                    // padded row stride in floats
#define WSLICE (NCAPS * DDIM * WROW)  // 18432 floats = 72 KB

// ---------------------------------------------------------------------------
// Stage 1: u_hat GEMM. One block per i; 4 waves, wave w owns b-rows [16w,16w+16).
// TDM stages W[:, i, :, :] (64 KB fp32) into LDS with 16B pad per 128B row.
// Per n: two 16x16 output tiles (d-tiles), each one v_wmma_f32_16x16x32_bf16.
// ---------------------------------------------------------------------------
__global__ __launch_bounds__(128)
void uhat_gemm_kernel(const float* __restrict__ x,
                      const float* __restrict__ W,
                      __bf16* __restrict__ uhat)
{
    __shared__ float lds_w[WSLICE];

    const int i    = blockIdx.x;
    const int wave = threadIdx.x >> 5;
    const int lane = threadIdx.x & 31;
    const int m    = lane & 15;     // row within A tile / col within B,C tiles
    const int h    = lane >> 4;     // half-wave select

    // ---- TDM: async-copy the W slice for this i into LDS (wave 0 issues) ----
    if (wave == 0) {
        const unsigned long long gaddr =
            (unsigned long long)(const char*)(W + (size_t)i * DDIM * KDIM);
        const unsigned int ldsoff = (unsigned int)(uintptr_t)(&lds_w[0]);

        // D# group 0: count=1 | lds_addr | global_addr[56:0] | type=2
        u32x4 g0;
        g0[0] = 1u;
        g0[1] = ldsoff;
        g0[2] = (unsigned int)gaddr;
        g0[3] = ((unsigned int)(gaddr >> 32) & 0x1FFFFFFu) | (2u << 30);

        // D# group 1: data_size=2 (4B), pad_enable, pad every 32 DW by 4 DW,
        // tensor_dim0=1024, tensor_dim1=16, tile_dim0=1024, tile_dim1=16,
        // tensor_dim0_stride = I*D*K = 2097152 elements.
        i32x8 g1;
        g1[0] = (2 << 16) | (1 << 20) | (4 << 22) | (3 << 25);
        g1[1] = (1024 << 16);          // tensor_dim0[15:0]
        g1[2] = (16 << 16);            // tensor_dim1[15:0]
        g1[3] = (1024 << 16);          // tile_dim0
        g1[4] = 16;                    // tile_dim1 (tile_dim2 = 0)
        g1[5] = (int)(ICAPS * DDIM * KDIM); // tensor_dim0_stride[31:0]
        g1[6] = 0;
        g1[7] = 0;

        i32x4 gz4 = {0, 0, 0, 0};      // 2-D tensor: groups 2/3 unused
        i32x8 gz8 = {0, 0, 0, 0, 0, 0, 0, 0};
        __builtin_amdgcn_tensor_load_to_lds(g0, g1, gz4, gz4, gz8, 0);
    }

    // ---- A fragment: 16x32 bf16 tile of x[b, i, :] for b = wave*16 + m ----
    // 16-bit A layout (wave32): lane half h holds K in [8h,8h+8) U [16+8h,16+8h+8)
    const int brow = wave * 16 + m;
    const float* xp = x + ((size_t)brow * ICAPS + i) * KDIM;
    float af[16];
    *(float4*)(af + 0)  = *(const float4*)(xp + h * 8);
    *(float4*)(af + 4)  = *(const float4*)(xp + h * 8 + 4);
    *(float4*)(af + 8)  = *(const float4*)(xp + 16 + h * 8);
    *(float4*)(af + 12) = *(const float4*)(xp + 16 + h * 8 + 4);
    v16bf A;
#pragma unroll
    for (int t = 0; t < 16; ++t) A[t] = (__bf16)af[t];

    if (wave == 0) __builtin_amdgcn_s_wait_tensorcnt(0);
    __syncthreads();

    for (int n = 0; n < NCAPS; ++n) {
#pragma unroll
        for (int tile = 0; tile < 2; ++tile) {
            // ---- B fragment from LDS: B[k, col] = W[n,i, d=tile*16+col, k]
            // 16-bit B layout: lane half h holds K in [8h,8h+8) U [16+8h,16+8h+8), col = lane%16.
            // Padded row stride (36 floats) spreads lanes across LDS banks.
            const float* wl = lds_w + ((n * DDIM) + tile * 16 + m) * WROW;
            float bf[16];
            *(float4*)(bf + 0)  = *(const float4*)(wl + h * 8);
            *(float4*)(bf + 4)  = *(const float4*)(wl + h * 8 + 4);
            *(float4*)(bf + 8)  = *(const float4*)(wl + 16 + h * 8);
            *(float4*)(bf + 12) = *(const float4*)(wl + 16 + h * 8 + 4);
            v16bf Bf;
#pragma unroll
            for (int t = 0; t < 16; ++t) Bf[t] = (__bf16)bf[t];

            v8f C = {};
            C = __builtin_amdgcn_wmma_f32_16x16x32_bf16(
                    /*neg_a=*/false, A, /*neg_b=*/false, Bf,
                    /*c_mod=*/(short)0, C, /*reuse_a=*/false, /*reuse_b=*/false);

            // ---- store C (16x16 f32 layout: VGPR r -> M = r + 8h, col = m) as bf16
#pragma unroll
            for (int r = 0; r < 8; ++r) {
                const int row = wave * 16 + r + 8 * h;  // batch index
                const size_t off = (((size_t)row * ICAPS + i) * NCAPS + n) * DDIM + tile * 16 + m;
                uhat[off] = (__bf16)C[r];
            }
        }
    }
}

// ---------------------------------------------------------------------------
// Stage 2: fused routing pass. grid = (B, I/64), block = 256 (8 waves).
// One wave handles one i at a time; lane = d. All of {a, logit update,
// softmax over n, s-accumulation} are local to the (b,i) slice.
// ---------------------------------------------------------------------------
template <bool FIRST>
__global__ __launch_bounds__(256)
void routing_pass_kernel(const __bf16* __restrict__ uhat,
                         float* __restrict__ logits,       // [B, I, N]
                         const float* __restrict__ vprev,  // [B, N, D]
                         float* __restrict__ sout)         // [B, N, D]
{
    __shared__ float s_l[NCAPS * DDIM];
    const int b    = blockIdx.x;
    const int wave = threadIdx.x >> 5;
    const int lane = threadIdx.x & 31;   // = d

    s_l[threadIdx.x]       = 0.f;
    s_l[threadIdx.x + 256] = 0.f;
    __syncthreads();

    float vv[NCAPS];
    if (!FIRST) {
#pragma unroll
        for (int n = 0; n < NCAPS; ++n)
            vv[n] = vprev[((size_t)b * NCAPS + n) * DDIM + lane];
    }

    float s_acc[NCAPS];
#pragma unroll
    for (int n = 0; n < NCAPS; ++n) s_acc[n] = 0.f;

    for (int it = 0; it < 8; ++it) {
        const int i = blockIdx.y * 64 + it * 8 + wave;
        const __bf16* up = uhat + (((size_t)b * ICAPS + i) * NCAPS) * DDIM + lane;

        float u[NCAPS];
#pragma unroll
        for (int n = 0; n < NCAPS; ++n) u[n] = (float)up[n * DDIM];

        float c[NCAPS];
        if (FIRST) {
#pragma unroll
            for (int n = 0; n < NCAPS; ++n) c[n] = 1.0f / NCAPS;
        } else {
            // a[n] = sum_d u[n,d] * v[n,d]  (butterfly reduce over 32 lanes)
            float a[NCAPS];
#pragma unroll
            for (int n = 0; n < NCAPS; ++n) a[n] = u[n] * vv[n];
#pragma unroll
            for (int off = 16; off >= 1; off >>= 1) {
#pragma unroll
                for (int n = 0; n < NCAPS; ++n)
                    a[n] += __shfl_xor(a[n], off, 32);
            }
            const size_t lbase = ((size_t)b * ICAPS + i) * NCAPS;
            float bn[NCAPS];
#pragma unroll
            for (int n = 0; n < NCAPS; ++n) bn[n] = logits[lbase + n] + a[n];

            // write updated logits (lane n stores bn[n])
            float wb = bn[0];
#pragma unroll
            for (int n = 1; n < NCAPS; ++n) wb = (lane == n) ? bn[n] : wb;
            if (lane < NCAPS) logits[lbase + lane] = wb;

            // softmax over n
            float mx = bn[0];
#pragma unroll
            for (int n = 1; n < NCAPS; ++n) mx = fmaxf(mx, bn[n]);
            float sum = 0.f;
#pragma unroll
            for (int n = 0; n < NCAPS; ++n) { c[n] = __expf(bn[n] - mx); sum += c[n]; }
            const float inv = 1.0f / sum;
#pragma unroll
            for (int n = 0; n < NCAPS; ++n) c[n] *= inv;
        }

#pragma unroll
        for (int n = 0; n < NCAPS; ++n) s_acc[n] = fmaf(c[n], u[n], s_acc[n]);
    }

    // merge waves via LDS float atomics, then one global atomic per element
#pragma unroll
    for (int n = 0; n < NCAPS; ++n)
        unsafeAtomicAdd(&s_l[n * DDIM + lane], s_acc[n]);
    __syncthreads();

    unsafeAtomicAdd(&sout[(size_t)b * (NCAPS * DDIM) + threadIdx.x],       s_l[threadIdx.x]);
    unsafeAtomicAdd(&sout[(size_t)b * (NCAPS * DDIM) + threadIdx.x + 256], s_l[threadIdx.x + 256]);
}

// ---------------------------------------------------------------------------
// Stage 3: squash. grid = B, block = 512 (wave = one n, lane = d).
// sq = sum_d(s^2 + eps);  v = sq*s / ((1+sq)*sqrt(sq))
// ---------------------------------------------------------------------------
__global__ __launch_bounds__(512)
void squash_kernel(const float* __restrict__ s, float* __restrict__ vout)
{
    const int b = blockIdx.x;
    const int t = threadIdx.x;
    const float sv = s[(size_t)b * (NCAPS * DDIM) + t];
    float sq = sv * sv;
#pragma unroll
    for (int off = 16; off >= 1; off >>= 1) sq += __shfl_xor(sq, off, 32);
    sq += (float)DDIM * EPSF;
    const float out = sq * sv / ((1.0f + sq) * sqrtf(sq));
    vout[(size_t)b * (NCAPS * DDIM) + t] = out;
}

// ---------------------------------------------------------------------------
extern "C" void kernel_launch(void* const* d_in, const int* in_sizes, int n_in,
                              void* d_out, int out_size, void* d_ws, size_t ws_size,
                              hipStream_t stream)
{
    const float* x = (const float*)d_in[0];  // [B, I, K]
    const float* W = (const float*)d_in[1];  // [N, I, D, K]

    // workspace layout: u_hat bf16 (128 MB) | logits f32 (8 MB) | s | vbuf
    const size_t UHAT_ELEMS = (size_t)BATCH * ICAPS * NCAPS * DDIM;
    __bf16* uhat  = (__bf16*)d_ws;
    float* logits = (float*)((char*)d_ws + UHAT_ELEMS * sizeof(__bf16));
    float* s      = logits + (size_t)BATCH * ICAPS * NCAPS;
    float* vbuf   = s + (size_t)BATCH * NCAPS * DDIM;

    (void)hipMemsetAsync(logits, 0, (size_t)BATCH * ICAPS * NCAPS * sizeof(float), stream);

    uhat_gemm_kernel<<<ICAPS, 128, 0, stream>>>(x, W, uhat);

    for (int r = 0; r < 3; ++r) {
        (void)hipMemsetAsync(s, 0, (size_t)BATCH * NCAPS * DDIM * sizeof(float), stream);
        if (r == 0)
            routing_pass_kernel<true><<<dim3(BATCH, ICAPS / 64), 256, 0, stream>>>(uhat, logits, vbuf, s);
        else
            routing_pass_kernel<false><<<dim3(BATCH, ICAPS / 64), 256, 0, stream>>>(uhat, logits, vbuf, s);
        squash_kernel<<<BATCH, 512, 0, stream>>>(s, (r == 2) ? (float*)d_out : vbuf);
    }
}
